// SortPool_73847667687835
// MI455X (gfx1250) — matrix operand order repaired and checked by reading, not compiled, not run on hardware
//
#include <hip/hip_runtime.h>
#include <hip/hip_bf16.h>

#define NN   100000
#define FIN  128
#define HID  64
#define EE   1600000
#define GG   1000
#define PER  100
#define KTOP 30

typedef __attribute__((ext_vector_type(2)))  float    v2f;
typedef __attribute__((ext_vector_type(8)))  float    v8f;
typedef __attribute__((ext_vector_type(16))) _Float16 v16h;

#if defined(__has_builtin)
#if __has_builtin(__builtin_amdgcn_wmma_f32_16x16x4_f32)
#define HAVE_WMMA_F32X4 1
#endif
#endif

static __device__ __forceinline__ int gtid() {
    return blockIdx.x * blockDim.x + threadIdx.x;
}

// ---------------- zero / count / reciprocal ----------------

__global__ void zero_kernel(float* p, int n) {
    int t = gtid();
    if (t < n) p[t] = 0.0f;
}

__global__ void count_kernel(const int* __restrict__ dst, float* __restrict__ cnt) {
    int t = gtid();
    if (t < EE) atomicAdd(&cnt[dst[t]], 1.0f);
}

__global__ void inv_kernel(float* cnt) {
    int t = gtid();
    if (t < NN) cnt[t] = 1.0f / fmaxf(cnt[t], 1.0f);
}

// ---------------- edge scatter: msum[dst] += x[src] ----------------

template <int F>
__global__ void scatter_kernel(const float* __restrict__ xin,
                               const int* __restrict__ src,
                               const int* __restrict__ dst,
                               float* __restrict__ msum) {
    const int C4 = F / 4;
    int t = gtid();
    if (t >= EE * C4) return;
    int e  = t / C4;
    int c4 = t % C4;
    int s = src[e];
    int d = dst[e];
    const float4 v = ((const float4*)(xin + (size_t)s * F))[c4];
    float* p = msum + (size_t)d * F + c4 * 4;
    atomicAdd(p + 0, v.x);
    atomicAdd(p + 1, v.y);
    atomicAdd(p + 2, v.z);
    atomicAdd(p + 3, v.w);
}

// ---------------- fused mean + dual GEMM + bias + relu -----------------
// out[N,64] = relu( (msum*invcnt) @ wl + x @ wr + b )
// 128 threads = 4 wave32; block computes 16 rows; wave w owns cols [16w,16w+16)

template <int F>
__global__ void __launch_bounds__(128)
sage_gemm_kernel(const float* __restrict__ msum,
                 const float* __restrict__ invcnt,
                 const float* __restrict__ xin,
                 const float* __restrict__ wl,
                 const float* __restrict__ wr,
                 const float* __restrict__ bias,
                 float* __restrict__ out) {
    __shared__ float sMean[16][F];
    __shared__ float sX[16][F];

    const int tid  = threadIdx.x;
    const int row0 = blockIdx.x * 16;

    for (int idx = tid; idx < 16 * F; idx += 128) {
        int r = idx / F, c = idx % F;
        int g = row0 + r;
        float ic = invcnt[g];
        sMean[r][c] = msum[(size_t)g * F + c] * ic;
        sX[r][c]    = xin[(size_t)g * F + c];
    }
    __syncthreads();

    const int lane = tid & 31;
    const int wave = tid >> 5;
    const int n0   = wave * 16;
    const int hi   = lane >> 4;    // lane half
    const int mrow = lane & 15;    // A row for this lane
    const int ncol = lane & 15;    // B/C/D column for this lane

    v8f acc = {};

#if HAVE_WMMA_F32X4
    // V_WMMA_F32_16X16X4_F32; A f32 16x4: element p of v2f holds K = k + p + 2*hi
    for (int k = 0; k < F; k += 4) {
        v2f a, b;
        a.x = sMean[mrow][k + 2 * hi];
        a.y = sMean[mrow][k + 1 + 2 * hi];
        b.x = wl[(size_t)(k + 2 * hi) * 64 + n0 + ncol];
        b.y = wl[(size_t)(k + 1 + 2 * hi) * 64 + n0 + ncol];
        acc = __builtin_amdgcn_wmma_f32_16x16x4_f32(false, a, false, b,
                                                    (short)0, acc, false, false);
    }
    for (int k = 0; k < F; k += 4) {
        v2f a, b;
        a.x = sX[mrow][k + 2 * hi];
        a.y = sX[mrow][k + 1 + 2 * hi];
        b.x = wr[(size_t)(k + 2 * hi) * 64 + n0 + ncol];
        b.y = wr[(size_t)(k + 1 + 2 * hi) * 64 + n0 + ncol];
        acc = __builtin_amdgcn_wmma_f32_16x16x4_f32(false, a, false, b,
                                                    (short)0, acc, false, false);
    }
#else
    // Fallback: V_WMMA_F32_16X16X32_F16 (codegen-confirmed builtin).
    // A f16 16x32: element e (v=e/2,p=e%2): K = 16*(v/4) + 2*(v%4) + p + 8*hi
    // B f16 32x16: element e: K = 16*(e/8) + e%8 + 8*hi
    for (int k = 0; k < F; k += 32) {
        v16h a, b;
        #pragma unroll
        for (int e = 0; e < 16; ++e) {
            int v = e >> 1, p = e & 1;
            int ka = k + 16 * (v >> 2) + 2 * (v & 3) + p + 8 * hi;
            int kb = k + 16 * (e >> 3) + (e & 7) + 8 * hi;
            a[e] = (_Float16)sMean[mrow][ka];
            b[e] = (_Float16)wl[(size_t)kb * 64 + n0 + ncol];
        }
        acc = __builtin_amdgcn_wmma_f32_16x16x32_f16(false, a, false, b,
                                                     (short)0, acc, false, false);
    }
    for (int k = 0; k < F; k += 32) {
        v16h a, b;
        #pragma unroll
        for (int e = 0; e < 16; ++e) {
            int v = e >> 1, p = e & 1;
            int ka = k + 16 * (v >> 2) + 2 * (v & 3) + p + 8 * hi;
            int kb = k + 16 * (e >> 3) + (e & 7) + 8 * hi;
            a[e] = (_Float16)sX[mrow][ka];
            b[e] = (_Float16)wr[(size_t)kb * 64 + n0 + ncol];
        }
        acc = __builtin_amdgcn_wmma_f32_16x16x32_f16(false, a, false, b,
                                                     (short)0, acc, false, false);
    }
#endif

    const float bv = bias[n0 + ncol];
    #pragma unroll
    for (int v = 0; v < 8; ++v) {
        int m = v + 8 * hi;                 // C/D layout: VGPR v, lane-half hi
        float val = acc[v] + bv;
        out[(size_t)(row0 + m) * 64 + n0 + ncol] = fmaxf(val, 0.0f);
    }
}

// ---------------- sort-pool + conv1d + MLP head (one block per graph) -----

__global__ void __launch_bounds__(128)
sortpool_head_kernel(const float* __restrict__ h,   // [N,64]
                     const float* __restrict__ cw,  // [32,64,3]
                     const float* __restrict__ cb,  // [32]
                     const float* __restrict__ w1,  // [896,64]
                     const float* __restrict__ bb1, // [64]
                     const float* __restrict__ w2,  // [64,1]
                     const float* __restrict__ bb2, // [1]
                     float* __restrict__ out) {     // [G]
    __shared__ float vals[PER];
    __shared__ float xcs[64 * KTOP];   // [c][r] = transposed top-k features
    __shared__ float ybuf[32 * 28];    // conv output, flattened o*28+p
    __shared__ float y1[64];

    const int g   = blockIdx.x;
    const int tid = threadIdx.x;
    const size_t base = (size_t)g * PER * 64;

    for (int i = tid; i < PER; i += 128)
        vals[i] = h[base + (size_t)i * 64 + 63];
    __syncthreads();

    // stable descending rank == jnp.argsort(-v) position; ranks are a permutation
    if (tid < PER) {
        float v = vals[tid];
        int rank = 0;
        for (int j = 0; j < PER; ++j) {
            float u = vals[j];
            rank += (u > v) || (u == v && j < tid);
        }
        if (rank < KTOP) {
            for (int c = 0; c < 64; ++c)
                xcs[c * KTOP + rank] = h[base + (size_t)tid * 64 + c];
        }
    }
    __syncthreads();

    // conv1d: y[o,p] = cb[o] + sum_{c,kk} xc[c][p+kk] * cw[o][c][kk]
    for (int t = tid; t < 32 * 28; t += 128) {
        int o = t / 28, p = t % 28;
        float acc = cb[o];
        for (int c = 0; c < 64; ++c) {
            const float* wrow = cw + ((size_t)o * 64 + c) * 3;
            const float* xr   = xcs + c * KTOP + p;
            acc += xr[0] * wrow[0] + xr[1] * wrow[1] + xr[2] * wrow[2];
        }
        ybuf[t] = fmaxf(acc, 0.0f);
    }
    __syncthreads();

    // [896] @ w1[896,64] + bb1, relu
    for (int t = tid; t < 64; t += 128) {
        float acc = bb1[t];
        for (int i = 0; i < 896; ++i)
            acc += ybuf[i] * w1[(size_t)i * 64 + t];
        y1[t] = fmaxf(acc, 0.0f);
    }
    __syncthreads();

    if (tid == 0) {
        float acc = bb2[0];
        for (int c = 0; c < 64; ++c) acc += y1[c] * w2[c];
        out[g] = acc;
    }
}

// ---------------- launch ----------------

static inline int cdiv(long long a, long long b) { return (int)((a + b - 1) / b); }

extern "C" void kernel_launch(void* const* d_in, const int* in_sizes, int n_in,
                              void* d_out, int out_size, void* d_ws, size_t ws_size,
                              hipStream_t stream) {
    const float* x   = (const float*)d_in[0];
    const int*   src = (const int*)d_in[1];
    const int*   dst = (const int*)d_in[2];
    const float* wl1 = (const float*)d_in[4];
    const float* wr1 = (const float*)d_in[5];
    const float* b1  = (const float*)d_in[6];
    const float* wl2 = (const float*)d_in[7];
    const float* wr2 = (const float*)d_in[8];
    const float* b2  = (const float*)d_in[9];
    const float* wl3 = (const float*)d_in[10];
    const float* wr3 = (const float*)d_in[11];
    const float* b3  = (const float*)d_in[12];
    const float* cw  = (const float*)d_in[13];
    const float* cb  = (const float*)d_in[14];
    const float* w1  = (const float*)d_in[15];
    const float* bb1 = (const float*)d_in[16];
    const float* w2  = (const float*)d_in[17];
    const float* bb2 = (const float*)d_in[18];
    float* out = (float*)d_out;

    float* ws   = (float*)d_ws;
    float* cnt  = ws;                       // NN floats (becomes invcnt)
    float* msum = cnt + 102400;             // NN*FIN floats
    float* hA   = msum + (size_t)NN * FIN;  // NN*HID
    float* hB   = hA + (size_t)NN * HID;    // NN*HID

    const int T = 256;

    // degree counts -> reciprocals
    zero_kernel<<<cdiv(NN, T), T, 0, stream>>>(cnt, NN);
    count_kernel<<<cdiv(EE, T), T, 0, stream>>>(dst, cnt);
    inv_kernel<<<cdiv(NN, T), T, 0, stream>>>(cnt);

    // layer 1: F=128 -> hA
    zero_kernel<<<cdiv((long long)NN * FIN, T), T, 0, stream>>>(msum, NN * FIN);
    scatter_kernel<FIN><<<cdiv((long long)EE * (FIN / 4), T), T, 0, stream>>>(x, src, dst, msum);
    sage_gemm_kernel<FIN><<<NN / 16, 128, 0, stream>>>(msum, cnt, x, wl1, wr1, b1, hA);

    // layer 2: F=64 -> hB
    zero_kernel<<<cdiv((long long)NN * HID, T), T, 0, stream>>>(msum, NN * HID);
    scatter_kernel<HID><<<cdiv((long long)EE * (HID / 4), T), T, 0, stream>>>(hA, src, dst, msum);
    sage_gemm_kernel<HID><<<NN / 16, 128, 0, stream>>>(msum, cnt, hA, wl2, wr2, b2, hB);

    // layer 3: F=64 -> hA
    zero_kernel<<<cdiv((long long)NN * HID, T), T, 0, stream>>>(msum, NN * HID);
    scatter_kernel<HID><<<cdiv((long long)EE * (HID / 4), T), T, 0, stream>>>(hB, src, dst, msum);
    sage_gemm_kernel<HID><<<NN / 16, 128, 0, stream>>>(msum, cnt, hB, wl3, wr3, b3, hA);

    // sort-pool + conv + MLP head
    sortpool_head_kernel<<<GG, 128, 0, stream>>>(hA, cw, cb, w1, bb1, w2, bb2, out);
}
